// SlidingWindowAttention_67130338836484
// MI455X (gfx1250) — matrix operand order repaired and checked by reading, not compiled, not run on hardware
//
#include <hip/hip_runtime.h>

// ---------------------------------------------------------------------------
// Fused Swin window attention for MI455X (gfx1250, wave32, WMMA).
// One workgroup (256 threads = 8 waves) processes one 7x7 window end-to-end
// in LDS: gather -> QKV GEMM -> attention (softmax) -> proj GEMM -> scatter.
// All matmuls use v_wmma_f32_16x16x32_f16 (f16 in, f32 accumulate).
// v2: double-buffered weight staging (1 barrier/K-step, global loads overlap
//     WMMA) and batched B-fragment loads (partial dscnt waits, fed pipe).
// ---------------------------------------------------------------------------

#define DIM      384
#define HEADS    12
#define WS       7
#define N_TOK    49          // WS*WS
#define NPAD     64          // tokens padded to 4 row-tiles of 16
#define HD       32          // head dim (= one WMMA K-step)
#define QKV_OUT  1152
#define WBLK     (192 * 32)  // staged weight block, f16 elems

typedef _Float16 h8   __attribute__((ext_vector_type(8)));
typedef _Float16 v16h __attribute__((ext_vector_type(16)));
typedef float    v8f  __attribute__((ext_vector_type(8)));

// LDS layout (bytes). Total 237,568 B of the 320 KB WGP LDS.
#define OFF_LX   0            // 64x384 f16   = 49152 (x window; reused as attn out)
#define OFF_LW   49152        // 2x 192x32 f16 = 24576 (double-buffered weights [n][k])
#define OFF_LQ   73728        // 12x64x32 f16 = 49152 (q, pre-scaled)
#define OFF_LK   122880       // 12x64x32 f16 = 49152 (k, row-major per key)
#define OFF_LV   172032       // 12x32x64 f16 = 49152 (v transposed: [hd][key])
#define OFF_LP   221184       // 8x16x64 f16  = 16384 (per-wave softmax probs)
#define SMEM_BYTES 237568
#define OFF_LO32 73728        // 64x384 f32 = 98304, aliases LQ+LK (dead by proj)

// ---- per-lane WMMA fragment loads, per CDNA5 ISA layouts -------
// A (16x32 f16): lanes 0-15 rows M=0..15 K in {kb..kb+7, kb+16..kb+23},
//                lanes 16-31 same rows, K in {kb+8..kb+15, kb+24..kb+31}.
__device__ __forceinline__ v16h ldsA(const _Float16* p, int stride, int mrow,
                                     int kb, int l15, int lhalf) {
  const _Float16* r = p + (mrow + l15) * stride + kb + lhalf * 8;
  h8 lo = *(const h8*)(r);
  h8 hi = *(const h8*)(r + 16);
  v16h o;
#pragma unroll
  for (int i = 0; i < 8; ++i) { o[i] = lo[i]; o[i + 8] = hi[i]; }
  return o;
}
// B (32x16 f16): lanes 0-15 col N=0..15 K in {kb..kb+7, kb+16..kb+23}, lanes 16-31 K in {kb+8..kb+15, kb+24..kb+31}.
// Source matrix stored "n-major": row = output column, contiguous K.
__device__ __forceinline__ v16h ldsB(const _Float16* p, int stride, int ncol,
                                     int kb, int l15, int lhalf) {
  const _Float16* r = p + (ncol + l15) * stride + kb + lhalf * 8;
  h8 lo = *(const h8*)(r);
  h8 hi = *(const h8*)(r + 16);
  v16h o;
#pragma unroll
  for (int i = 0; i < 8; ++i) { o[i] = lo[i]; o[i + 8] = hi[i]; }
  return o;
}

__device__ __forceinline__ v8f wmma16(v16h a, v16h b, v8f c) {
  return __builtin_amdgcn_wmma_f32_16x16x32_f16(false, a, false, b, (short)0, c,
                                                false, false);
}

// stage a 32x192 f32 weight block, transposed into dst[n][k] as f16
__device__ __forceinline__ void stageW(_Float16* dst, const float* __restrict__ src,
                                       int ks, int nbase, int ld, int tid) {
  for (int e = tid; e < WBLK; e += 256) {
    int n = e % 192, k = e / 192;     // n fastest -> coalesced global reads
    dst[n * 32 + k] = (_Float16)src[(ks * 32 + k) * ld + nbase + n];
  }
}

__global__ __launch_bounds__(256)
void swin_window_attn_kernel(const float* __restrict__ x,
                             const float* __restrict__ qkv_w,
                             const float* __restrict__ qkv_b,
                             const float* __restrict__ proj_w,
                             const float* __restrict__ proj_b,
                             const float* __restrict__ bias_table,
                             const int*   __restrict__ rel_index,
                             float* __restrict__ out) {
  __shared__ __align__(16) char smem[SMEM_BYTES];
  _Float16* lx   = (_Float16*)(smem + OFF_LX);
  _Float16* lw   = (_Float16*)(smem + OFF_LW);
  _Float16* lq   = (_Float16*)(smem + OFF_LQ);
  _Float16* lk   = (_Float16*)(smem + OFF_LK);
  _Float16* lv   = (_Float16*)(smem + OFF_LV);
  _Float16* lp   = (_Float16*)(smem + OFF_LP);
  float*    lo32 = (float*)(smem + OFF_LO32);

  const int tid   = threadIdx.x;
  const int wave  = tid >> 5;
  const int lane  = tid & 31;
  const int l15   = lane & 15;
  const int lhalf = lane >> 4;

  const int bid = blockIdx.x;           // b*64 + wy*8 + wx
  const int b   = bid >> 6;
  const int wy  = (bid >> 3) & 7;
  const int wx  = bid & 7;
  const int gy0 = wy * WS, gx0 = wx * WS;

  // ---- Phase 0: gather window x[b,:,gy0:gy0+7,gx0:gx0+7] -> lx (f16) ------
  for (int e = tid; e < NPAD * DIM; e += 256) {
    int n = e & 63;                     // token fastest -> 7-float global runs
    int c = e >> 6;
    _Float16 v = (_Float16)0.f;
    if (n < N_TOK) {
      int iy = n / WS, ix = n - iy * WS;
      v = (_Float16)x[(((long)b * DIM + c) * 56 + gy0 + iy) * 56 + gx0 + ix];
    }
    lx[n * DIM + c] = v;
  }

  const int  mt  = wave & 3;            // row tile owned in GEMM phases
  const int  nhf = wave >> 2;           // column half (0/1)
  const float qscale = 0.17677669529663687f;   // 1/sqrt(hd)
  v8f zf = {};

  // ---- Phase A: qkv = lx(64x384) @ qkv_w(384x1152) + qkv_b ----------------
  for (int nb = 0; nb < 6; ++nb) {      // 6 blocks of 192 output cols
    v8f acc[6] = {};
    stageW(lw, qkv_w, 0, nb * 192, QKV_OUT, tid);   // prologue: buf 0
    __syncthreads();
    for (int ks = 0; ks < 12; ++ks) {   // 12 K-steps of 32
      const _Float16* cur = lw + (ks & 1) * WBLK;
      if (ks + 1 < 12)                  // stage next block into other buffer
        stageW(lw + ((ks + 1) & 1) * WBLK, qkv_w, ks + 1, nb * 192, QKV_OUT, tid);
      v16h a = ldsA(lx, DIM, mt * 16, ks * 32, l15, lhalf);
      v16h bf[6];
#pragma unroll
      for (int i = 0; i < 6; ++i)
        bf[i] = ldsB(cur, 32, (nhf * 6 + i) * 16, 0, l15, lhalf);
#pragma unroll
      for (int i = 0; i < 6; ++i) acc[i] = wmma16(a, bf[i], acc[i]);
      __syncthreads();
    }
    // epilogue: +bias, scatter into q (scaled) / k / v (transposed) layouts
#pragma unroll
    for (int i = 0; i < 6; ++i) {
      int col = nb * 192 + (nhf * 6 + i) * 16 + l15;    // 0..1151
      float bv = qkv_b[col];
#pragma unroll
      for (int r = 0; r < 8; ++r) {
        int row = mt * 16 + lhalf * 8 + r;
        float v = acc[i][r] + bv;
        if (col < DIM) {
          int h = col >> 5, c = col & 31;
          lq[(h * NPAD + row) * HD + c] = (_Float16)(v * qscale);
        } else if (col < 2 * DIM) {
          int cc = col - DIM, h = cc >> 5, c = cc & 31;
          lk[(h * NPAD + row) * HD + c] = (_Float16)v;
        } else {
          int cc = col - 2 * DIM, h = cc >> 5, c = cc & 31;
          lv[(h * HD + c) * NPAD + row] = (_Float16)v;  // [hd][key]
        }
      }
    }
  }
  __syncthreads();

  // ---- Phase B: per (head, 16-row tile): scores -> softmax -> P@V ---------
  _Float16* lpw = lp + wave * 16 * 64;  // private per-wave prob scratch
  for (int u = wave; u < HEADS * 4; u += 8) {
    int h = u >> 2, mtl = u & 3;
    const _Float16* qh = lq + h * NPAD * HD;
    const _Float16* kh = lk + h * NPAD * HD;
    const _Float16* vh = lv + h * HD * NPAD;
    v16h aq = ldsA(qh, HD, mtl * 16, 0, l15, lhalf);
    v16h bk[4];
#pragma unroll
    for (int nt = 0; nt < 4; ++nt)
      bk[nt] = ldsB(kh, HD, nt * 16, 0, l15, lhalf);
    v8f s[4];
#pragma unroll
    for (int nt = 0; nt < 4; ++nt)      // q(16x32) @ k^T(32x16): K=hd, 1 WMMA
      s[nt] = wmma16(aq, bk[nt], zf);
    // relative-position bias + key mask (cols >= 49)
    float p[4][8];
#pragma unroll
    for (int nt = 0; nt < 4; ++nt) {
      int nc = nt * 16 + l15;
#pragma unroll
      for (int r = 0; r < 8; ++r) {
        int mr = mtl * 16 + lhalf * 8 + r;
        float v = -1e30f;
        if (nc < N_TOK && mr < N_TOK)
          v = s[nt][r] + bias_table[rel_index[mr * N_TOK + nc] * HEADS + h];
        p[nt][r] = v;
      }
    }
    // softmax over 64 key slots = 4 tiles x 16 lanes (reduce within half-wave)
#pragma unroll
    for (int r = 0; r < 8; ++r) {
      float m = fmaxf(fmaxf(p[0][r], p[1][r]), fmaxf(p[2][r], p[3][r]));
      m = fmaxf(m, __shfl_xor(m, 1, 16));
      m = fmaxf(m, __shfl_xor(m, 2, 16));
      m = fmaxf(m, __shfl_xor(m, 4, 16));
      m = fmaxf(m, __shfl_xor(m, 8, 16));
      float sum = 0.f;
#pragma unroll
      for (int nt = 0; nt < 4; ++nt) { p[nt][r] = __expf(p[nt][r] - m); sum += p[nt][r]; }
      sum += __shfl_xor(sum, 1, 16);
      sum += __shfl_xor(sum, 2, 16);
      sum += __shfl_xor(sum, 4, 16);
      sum += __shfl_xor(sum, 8, 16);
      float inv = 1.f / sum;
#pragma unroll
      for (int nt = 0; nt < 4; ++nt) p[nt][r] *= inv;
    }
    // stash P (16x64, row-major f16) so it can be re-read as a WMMA A operand
#pragma unroll
    for (int nt = 0; nt < 4; ++nt)
#pragma unroll
      for (int r = 0; r < 8; ++r)
        lpw[(lhalf * 8 + r) * 64 + nt * 16 + l15] = (_Float16)p[nt][r];
    // out_tile(16x32) = P(16x64) @ V(64x32): 2 K-steps x 2 col tiles
    v16h ap[2], bv[2][2];
#pragma unroll
    for (int ksp = 0; ksp < 2; ++ksp) {
      ap[ksp] = ldsA(lpw, 64, 0, ksp * 32, l15, lhalf);
#pragma unroll
      for (int nt2 = 0; nt2 < 2; ++nt2)
        bv[ksp][nt2] = ldsB(vh, NPAD, nt2 * 16, ksp * 32, l15, lhalf);
    }
    v8f o[2] = {};
#pragma unroll
    for (int ksp = 0; ksp < 2; ++ksp)
#pragma unroll
      for (int nt2 = 0; nt2 < 2; ++nt2)
        o[nt2] = wmma16(ap[ksp], bv[ksp][nt2], o[nt2]);
#pragma unroll
    for (int nt2 = 0; nt2 < 2; ++nt2)
#pragma unroll
      for (int r = 0; r < 8; ++r)
        lx[(mtl * 16 + lhalf * 8 + r) * DIM + h * HD + nt2 * 16 + l15] =
            (_Float16)o[nt2][r];
  }
  __syncthreads();

  // ---- Phase C: out = lx(64x384) @ proj_w(384x384) + proj_b ---------------
  for (int nb = 0; nb < 2; ++nb) {
    v8f acc[6] = {};
    stageW(lw, proj_w, 0, nb * 192, DIM, tid);
    __syncthreads();
    for (int ks = 0; ks < 12; ++ks) {
      const _Float16* cur = lw + (ks & 1) * WBLK;
      if (ks + 1 < 12)
        stageW(lw + ((ks + 1) & 1) * WBLK, proj_w, ks + 1, nb * 192, DIM, tid);
      v16h a = ldsA(lx, DIM, mt * 16, ks * 32, l15, lhalf);
      v16h bf[6];
#pragma unroll
      for (int i = 0; i < 6; ++i)
        bf[i] = ldsB(cur, 32, (nhf * 6 + i) * 16, 0, l15, lhalf);
#pragma unroll
      for (int i = 0; i < 6; ++i) acc[i] = wmma16(a, bf[i], acc[i]);
      __syncthreads();
    }
#pragma unroll
    for (int i = 0; i < 6; ++i) {
      int col = nb * 192 + (nhf * 6 + i) * 16 + l15;    // 0..383
      float bv = proj_b[col];
#pragma unroll
      for (int r = 0; r < 8; ++r) {
        int row = mt * 16 + lhalf * 8 + r;
        lo32[row * DIM + col] = acc[i][r] + bv;         // f32, aliases lq/lk
      }
    }
  }
  __syncthreads();

  // ---- writeout: NCHW scatter, token-fastest -> contiguous 7-float runs ---
  for (int e = tid; e < N_TOK * DIM; e += 256) {
    int n = e % N_TOK, c = e / N_TOK;
    int iy = n / WS, ix = n - iy * WS;
    out[(((long)b * DIM + c) * 56 + gy0 + iy) * 56 + gx0 + ix] = lo32[n * DIM + c];
  }
}

extern "C" void kernel_launch(void* const* d_in, const int* in_sizes, int n_in,
                              void* d_out, int out_size, void* d_ws, size_t ws_size,
                              hipStream_t stream) {
  const float* x      = (const float*)d_in[0];
  const float* qkv_w  = (const float*)d_in[1];
  const float* qkv_b  = (const float*)d_in[2];
  const float* proj_w = (const float*)d_in[3];
  const float* proj_b = (const float*)d_in[4];
  const float* btab   = (const float*)d_in[5];
  const int*   relidx = (const int*)d_in[6];
  float* out = (float*)d_out;

  const int B = in_sizes[0] / (DIM * 56 * 56);     // 32
  dim3 grid(B * 64), block(256);                   // one workgroup per window
  hipLaunchKernelGGL(swin_window_attn_kernel, grid, block, 0, stream,
                     x, qkv_w, qkv_b, proj_w, proj_b, btab, relidx, out);
}
